// MultiHeadAttention_77833397338219
// MI455X (gfx1250) — matrix operand order repaired and checked by reading, not compiled, not run on hardware
//
#include <hip/hip_runtime.h>
#include <hip/hip_bf16.h>

// ---------------------------------------------------------------------------
// MHA forward for MI455X (gfx1250): bf16 WMMA everywhere, flash-attention
// streaming (S^T formulation), f32 accumulate. Double-buffered GEMMs using
// GLOBAL_LOAD_ASYNC_TO_LDS (ASYNCcnt) when the toolchain exposes it.
//   B=2, T=2048, C=768, H=12, D=64.  M = B*T = 4096.
// Workspace (bytes, 256-aligned; ~34.5 MB total):
//   xb @0, wat @6291456, wpt @9830400, qw @11010048, kw @17301504,
//   vt @23592960, yb @29884416
// ---------------------------------------------------------------------------

typedef __bf16 bf16_t;
typedef __attribute__((ext_vector_type(16))) __bf16 v16bf;
typedef __attribute__((ext_vector_type(8)))  __bf16 v8bf;
typedef __attribute__((ext_vector_type(8)))  float  v8f;

#define WMMA_BF16(A, B, C) \
  __builtin_amdgcn_wmma_f32_16x16x32_bf16(false, (A), false, (B), (short)0, (C), false, false)

#if defined(__has_builtin)
#  if __has_builtin(__builtin_amdgcn_global_load_async_to_lds_b128)
#    define HAVE_ASYNC_B128 1
#  endif
#  if __has_builtin(__builtin_amdgcn_s_wait_asynccnt)
#    define HAVE_WAIT_ASYNC 1
#  endif
#endif

#if defined(HAVE_ASYNC_B128)
typedef __attribute__((__vector_size__(4 * sizeof(int)))) int v4i_t;
typedef __attribute__((address_space(1))) v4i_t gbl_v4i_t;   // global segment
typedef __attribute__((address_space(3))) v4i_t lds_v4i_t;   // LDS segment
__device__ __forceinline__ void async_b128(const void* g, void* l) {
  __builtin_amdgcn_global_load_async_to_lds_b128((gbl_v4i_t*)g, (lds_v4i_t*)l, 0, 0);
}
#  if defined(HAVE_WAIT_ASYNC)
#    define WAIT_ASYNC0() __builtin_amdgcn_s_wait_asynccnt(0)
#  else
#    define WAIT_ASYNC0() asm volatile("s_wait_asynccnt 0x0" ::: "memory")
#  endif
#endif

// ---------------------------------------------------------------- stage 0 ---
__global__ __launch_bounds__(256) void cvt_f32_bf16(const float* __restrict__ in,
                                                    bf16_t* __restrict__ out, int n) {
  int i = blockIdx.x * 256 + threadIdx.x;
  if (i < n) out[i] = (bf16_t)in[i];
}

// wt[n][k] = w[k][n]   (w is [K][N] row-major)
__global__ __launch_bounds__(256) void transpose_w(const float* __restrict__ w,
                                                   bf16_t* __restrict__ wt, int K, int N) {
  int i = blockIdx.x * 256 + threadIdx.x;
  if (i < K * N) {
    int n = i / K, k = i - n * K;
    wt[i] = (bf16_t)w[(size_t)k * N + n];
  }
}

// ----------------------------------------------------------- GEMM skeleton ---
// Block tile 64(M) x 128(N), 8 waves of 16x64, K-step 32, K total 768.
// Double-buffered LDS (flattened, runtime XOR-toggled offsets so the loop is
// NOT unrolled -> single accumulator phi cycle -> in-place WMMA accumulation).

#define LA_HALF (64 * 32)    // elements per A buffer half
#define LB_HALF (128 * 32)   // elements per B buffer half

#define GEMM_PROLOGUE(APTR, BPTR)                                              \
  __shared__ __align__(64) bf16_t la[2 * LA_HALF];                             \
  __shared__ __align__(64) bf16_t lb[2 * LB_HALF];                             \
  const int tid  = threadIdx.x;                                                \
  const int lane = tid & 31;                                                   \
  const int wave = tid >> 5;                                                   \
  const int wm   = (wave & 3) * 16;                                            \
  const int wn   = (wave >> 2) * 64;                                           \
  const int m0   = blockIdx.x * 64;                                            \
  const int n0   = blockIdx.y * 128;                                           \
  const int half = lane >> 4;                                                  \
  const int l16  = lane & 15;                                                  \
  v8f acc[4] = {};                                                             \
  const int ra = tid >> 2, ca = (tid & 3) * 8;  /* A: 8 bf16 / thread  */      \
  const int rb = tid >> 1, cb = (tid & 1) * 16; /* B: 16 bf16 / thread */      \
  const bf16_t* gA = &(APTR)[(size_t)(m0 + ra) * 768 + ca];                    \
  const bf16_t* gB = &(BPTR)[(size_t)(n0 + rb) * 768 + cb];

#define GEMM_COMPUTE_OFF(AO, BO)                                               \
  {                                                                            \
    v16bf af;                                                                  \
    {                                                                          \
      v8bf lo = *(const v8bf*)&la[(AO) + (wm + l16) * 32 + half * 16];         \
      v8bf hi = *(const v8bf*)&la[(AO) + (wm + l16) * 32 + 8 + half * 16];     \
      _Pragma("unroll")                                                        \
      for (int j = 0; j < 8; ++j) { af[j] = lo[j]; af[8 + j] = hi[j]; }        \
    }                                                                          \
    _Pragma("unroll")                                                          \
    for (int s = 0; s < 4; ++s) {                                              \
      v16bf bfrag =                                                            \
          *(const v16bf*)&lb[(BO) + (wn + s * 16 + l16) * 32 + half * 16];     \
      acc[s] = WMMA_BF16(af, bfrag, acc[s]);                                   \
    }                                                                          \
  }

#if defined(HAVE_ASYNC_B128)
#define GEMM_ISSUE_OFF(I, AO, BO)                                              \
  {                                                                            \
    const int kn = (I) * 32;                                                   \
    async_b128(gA + kn, &la[(AO) + ra * 32 + ca]);                             \
    async_b128(gB + kn, &lb[(BO) + rb * 32 + cb]);                             \
    async_b128(gB + kn + 8, &lb[(BO) + rb * 32 + cb + 8]);                     \
  }

#define GEMM_MAINLOOP()                                                        \
  GEMM_ISSUE_OFF(0, 0, 0)                                                      \
  int aoff = 0, boff = 0;                                                      \
  _Pragma("clang loop unroll(disable)")                                        \
  for (int i = 0; i < 23; ++i) {                                               \
    WAIT_ASYNC0();          /* this wave's slab-i loads landed in LDS */       \
    __syncthreads();        /* all waves landed; all done with slab i-1 */     \
    GEMM_ISSUE_OFF(i + 1, aoff ^ LA_HALF, boff ^ LB_HALF)                      \
    GEMM_COMPUTE_OFF(aoff, boff)                                               \
    aoff ^= LA_HALF;                                                           \
    boff ^= LB_HALF;                                                           \
  }                                                                            \
  WAIT_ASYNC0();                                                               \
  __syncthreads();                                                             \
  GEMM_COMPUTE_OFF(aoff, boff)   /* slab 23 (aoff==LA_HALF after 23 flips) */
#else
#define GEMM_MAINLOOP()                                                        \
  for (int k0 = 0; k0 < 768; k0 += 32) {                                       \
    *(v8bf*)&la[ra * 32 + ca]  = *(const v8bf*)(gA + k0);                      \
    *(v16bf*)&lb[rb * 32 + cb] = *(const v16bf*)(gB + k0);                     \
    if (k0 + 32 < 768) {                                                       \
      __builtin_prefetch(gA + k0 + 32, 0, 1);                                  \
      __builtin_prefetch(gB + k0 + 32, 0, 1);                                  \
    }                                                                          \
    __syncthreads();                                                           \
    GEMM_COMPUTE_OFF(0, 0)                                                     \
    __syncthreads();                                                           \
  }
#endif

// ---------------------------------------------------------------- stage 1 ---
// C[4096,2304] = xb @ wat^T + bias, scattered into q / k / v^T (bf16).
__global__ __launch_bounds__(256) void qkv_gemm(const bf16_t* __restrict__ xb,
                                                const bf16_t* __restrict__ wt,
                                                const float* __restrict__ bias,
                                                bf16_t* __restrict__ qw,
                                                bf16_t* __restrict__ kw,
                                                bf16_t* __restrict__ vt) {
  GEMM_PROLOGUE(xb, wt)
  GEMM_MAINLOOP()

#pragma unroll
  for (int s = 0; s < 4; ++s) {
    const int n     = n0 + wn + s * 16 + l16;        // 0..2303
    const int which = n / 768;
    const int rem   = n - which * 768;
    const int h     = rem >> 6;
    const int d     = rem & 63;
    const float bv  = bias[n];
#pragma unroll
    for (int r = 0; r < 8; ++r) {
      const int m  = m0 + wm + r + 8 * half;
      const int b  = m >> 11;
      const int t  = m & 2047;
      const int bh = b * 12 + h;
      const bf16_t o = (bf16_t)(acc[s][r] + bv);
      if (which == 0)      qw[((size_t)bh * 2048 + t) * 64 + d] = o;
      else if (which == 1) kw[((size_t)bh * 2048 + t) * 64 + d] = o;
      else                 vt[((size_t)bh * 64 + d) * 2048 + t] = o;
    }
  }
}

// ---------------------------------------------------------------- stage 3 ---
// out[4096,768] = yb @ wpt^T + b_proj   (f32 output)
__global__ __launch_bounds__(256) void proj_gemm(const bf16_t* __restrict__ yb,
                                                 const bf16_t* __restrict__ wt,
                                                 const float* __restrict__ bias,
                                                 float* __restrict__ out) {
  GEMM_PROLOGUE(yb, wt)
  GEMM_MAINLOOP()

#pragma unroll
  for (int s = 0; s < 4; ++s) {
    const int n    = n0 + wn + s * 16 + l16;
    const float bv = bias[n];
#pragma unroll
    for (int r = 0; r < 8; ++r) {
      const int m = m0 + wm + r + 8 * half;
      out[(size_t)m * 768 + n] = acc[s][r] + bv;
    }
  }
}

// ---------------------------------------------------------------- stage 2 ---
// Flash attention, one wave per (b,h,16-query tile). S^T = K Q^T formulation:
// softmax stats per query column live in one lane; half-wave shfl_xor(16)
// handles both the stat reduction and the C-layout -> B-operand transpose.
__global__ __launch_bounds__(32) void flash_attn(const bf16_t* __restrict__ qw,
                                                 const bf16_t* __restrict__ kw,
                                                 const bf16_t* __restrict__ vt,
                                                 bf16_t* __restrict__ yb) {
  const int lane = threadIdx.x & 31;
  const int half = lane >> 4, l16 = lane & 15;
  const int qt   = blockIdx.x & 127;   // query tile 0..127
  const int bh   = blockIdx.x >> 7;    // 0..23
  const int b    = bh / 12, h = bh - b * 12;
  const int q0   = qt * 16;

  const bf16_t* qp = qw + (size_t)bh * 2048 * 64;
  const bf16_t* kp = kw + (size_t)bh * 2048 * 64;
  const bf16_t* vp = vt + (size_t)bh * 64 * 2048;

  // Q^T B-fragments (kdim = d chunks of 32), resident for whole loop.
  const v16bf bq0 = *(const v16bf*)&qp[(size_t)(q0 + l16) * 64 + half * 16];
  const v16bf bq1 = *(const v16bf*)&qp[(size_t)(q0 + l16) * 64 + 32 + half * 16];

  v8f ot[4] = {};                       // O^T accumulators: 4 d-tiles x (16x16)
  float m_run = -__builtin_inff();
  float l_run = 0.f;
  const float sc = 0.125f * 1.44269504088896340736f;  // 1/sqrt(64) * log2(e)
  const int  qcol = q0 + l16;
  const int  nkt  = (qt >> 1) + 1;      // causal tile count (32 keys per tile)

  for (int kt = 0; kt < nkt; ++kt) {
    const int k0 = kt * 32;

    // S^T tiles (k rows x q cols), two 16x16 tiles over the 32-key slab.
    v8f st[2];
#pragma unroll
    for (int t2 = 0; t2 < 2; ++t2) {
      const bf16_t* kr = &kp[(size_t)(k0 + t2 * 16 + l16) * 64];
      v16bf a0, a1;
      {
        v8bf lo = *(const v8bf*)&kr[half * 16];
        v8bf hi = *(const v8bf*)&kr[8 + half * 16];
#pragma unroll
        for (int j = 0; j < 8; ++j) { a0[j] = lo[j]; a0[8 + j] = hi[j]; }
      }
      {
        v8bf lo = *(const v8bf*)&kr[32 + half * 16];
        v8bf hi = *(const v8bf*)&kr[40 + half * 16];
#pragma unroll
        for (int j = 0; j < 8; ++j) { a1[j] = lo[j]; a1[8 + j] = hi[j]; }
      }
      v8f z = {};
      z = WMMA_BF16(a0, bq0, z);
      z = WMMA_BF16(a1, bq1, z);
      st[t2] = z;
    }

    // Causal mask + scale (log2 domain), column (=lane) max.
    float p[2][8];
    float tmax = -__builtin_inff();
#pragma unroll
    for (int t2 = 0; t2 < 2; ++t2)
#pragma unroll
      for (int r = 0; r < 8; ++r) {
        const int krow = k0 + t2 * 16 + 8 * half + r;
        const float v  = (krow <= qcol) ? st[t2][r] * sc : -__builtin_inff();
        p[t2][r] = v;
        tmax = fmaxf(tmax, v);
      }
    tmax = fmaxf(tmax, __shfl_xor(tmax, 16, 32));
    const float nm    = fmaxf(m_run, tmax);
    const float alpha = __builtin_amdgcn_exp2f(m_run - nm);

    float lsum = 0.f;
#pragma unroll
    for (int t2 = 0; t2 < 2; ++t2)
#pragma unroll
      for (int r = 0; r < 8; ++r) {
        const float e = __builtin_amdgcn_exp2f(p[t2][r] - nm);
        p[t2][r] = e;
        lsum += e;
      }
    lsum += __shfl_xor(lsum, 16, 32);
    l_run = l_run * alpha + lsum * 0.5f;  // both halves summed the full column
    m_run = nm;

#pragma unroll
    for (int dt = 0; dt < 4; ++dt)
#pragma unroll
      for (int r = 0; r < 8; ++r) ot[dt][r] *= alpha;

    // P^T C-layout -> B-operand layout via one half-wave exchange.
    float x0[8], x1[8];
#pragma unroll
    for (int r = 0; r < 8; ++r) {
      x0[r] = __shfl_xor(p[0][r], 16, 32);
      x1[r] = __shfl_xor(p[1][r], 16, 32);
    }
    v16bf bp;
#pragma unroll
    for (int j = 0; j < 8; ++j) {
      const float lo = (half == 0) ? p[0][j] : x1[j];
      const float hi = (half == 0) ? x0[j]   : p[1][j];
      bp[j]     = (bf16_t)lo;
      bp[8 + j] = (bf16_t)hi;
    }

    // O^T += V^T P^T  (A = V^T fragments from pre-transposed V).
#pragma unroll
    for (int dt = 0; dt < 4; ++dt) {
      const bf16_t* vr = &vp[(size_t)(dt * 16 + l16) * 2048 + k0];
      v16bf av;
      v8bf lo = *(const v8bf*)&vr[half * 16];
      v8bf hi = *(const v8bf*)&vr[8 + half * 16];
#pragma unroll
      for (int j = 0; j < 8; ++j) { av[j] = lo[j]; av[8 + j] = hi[j]; }
      ot[dt] = WMMA_BF16(av, bp, ot[dt]);
    }
  }

  // Epilogue: normalize, write y row (t = q0 + lane column), bf16.
  const float rl = 1.f / l_run;
  const int t = q0 + l16;
  bf16_t* yrow = yb + (size_t)(b * 2048 + t) * 768 + h * 64;
#pragma unroll
  for (int dt = 0; dt < 4; ++dt) {
    v8bf o;
#pragma unroll
    for (int r = 0; r < 8; ++r) o[r] = (bf16_t)(ot[dt][r] * rl);
    *(v8bf*)&yrow[dt * 16 + 8 * half] = o;
  }
}

// ---------------------------------------------------------------------------
extern "C" void kernel_launch(void* const* d_in, const int* in_sizes, int n_in,
                              void* d_out, int out_size, void* d_ws, size_t ws_size,
                              hipStream_t stream) {
  const float* x      = (const float*)d_in[0];
  const float* w_attn = (const float*)d_in[1];
  const float* b_attn = (const float*)d_in[2];
  const float* w_proj = (const float*)d_in[3];
  const float* b_proj = (const float*)d_in[4];
  float* out = (float*)d_out;

  char* ws = (char*)d_ws;   // requires ~34.5 MB of workspace
  bf16_t* xb  = (bf16_t*)(ws + 0);
  bf16_t* wat = (bf16_t*)(ws + 6291456);
  bf16_t* wpt = (bf16_t*)(ws + 9830400);
  bf16_t* qw  = (bf16_t*)(ws + 11010048);
  bf16_t* kw  = (bf16_t*)(ws + 17301504);
  bf16_t* vt  = (bf16_t*)(ws + 23592960);
  bf16_t* yb  = (bf16_t*)(ws + 29884416);

  cvt_f32_bf16<<<(3145728 + 255) / 256, 256, 0, stream>>>(x, xb, 3145728);
  transpose_w<<<(1769472 + 255) / 256, 256, 0, stream>>>(w_attn, wat, 768, 2304);
  transpose_w<<<(589824 + 255) / 256, 256, 0, stream>>>(w_proj, wpt, 768, 768);
  qkv_gemm<<<dim3(64, 18), 256, 0, stream>>>(xb, wat, b_attn, qw, kw, vt);
  flash_attn<<<3072, 32, 0, stream>>>(qw, kw, vt, yb);
  proj_gemm<<<dim3(64, 6), 256, 0, stream>>>(yb, wpt, b_proj, out);
}
